// Model_4277787427274
// MI455X (gfx1250) — compile-verified, matrix operand order checked by reading, not run on hardware
//
#include <hip/hip_runtime.h>
#include <hip/hip_bf16.h>

typedef __attribute__((ext_vector_type(16))) _Float16     v16h;
typedef __attribute__((ext_vector_type(8)))  _Float16     v8h;
typedef __attribute__((ext_vector_type(8)))  float        v8f;
typedef __attribute__((ext_vector_type(8)))  unsigned int v8u;

#define BN_EPS 1e-5f

// ---------------------------------------------------------------------------
// Attention: per (head k, node i): c_n = tanh(emb_e @ W1 + b1) @ w2 + b2,
// softmax over nn neighbours. One wave (32 lanes) per (k,i); each lane owns
// 4 of the 128 hidden units, wave-reduce via shfl_xor (wave32!).
// ---------------------------------------------------------------------------
__global__ void attn_kernel(const float* __restrict__ emb,
                            const float* __restrict__ W1, const float* __restrict__ B1,
                            const float* __restrict__ W2, const float* __restrict__ B2,
                            float* __restrict__ aout, int m, int nn)
{
    const int k = blockIdx.x / m;
    const int i = blockIdx.x % m;
    const int lane = threadIdx.x;
    const float* w1 = W1 + (size_t)k * 5 * 128;
    const float* b1 = B1 + (size_t)k * 128;
    const float* w2 = W2 + (size_t)k * 128;
    const float  b2 = B2[k];

    float c[16];
#pragma unroll 16
    for (int n = 0; n < 16; ++n) {
        c[n] = -1e30f;
        if (n < nn) {
            const float* e5 = emb + (size_t)(i * nn + n) * 5;
            float e0 = e5[0], e1 = e5[1], e2 = e5[2], e3 = e5[3], e4 = e5[4];
            float part = 0.f;
#pragma unroll
            for (int q = 0; q < 4; ++q) {
                int hid = lane + 32 * q;
                float s = b1[hid] + e0 * w1[hid] + e1 * w1[128 + hid] + e2 * w1[256 + hid]
                                  + e3 * w1[384 + hid] + e4 * w1[512 + hid];
                part += tanhf(s) * w2[hid];
            }
#pragma unroll
            for (int o = 16; o > 0; o >>= 1) part += __shfl_xor(part, o, 32);
            c[n] = part + b2;
        }
    }
    float mx = -1e30f;
#pragma unroll 16
    for (int n = 0; n < 16; ++n) mx = fmaxf(mx, c[n]);
    float sum = 0.f;
#pragma unroll 16
    for (int n = 0; n < 16; ++n) {
        if (n < nn) { c[n] = __expf(c[n] - mx); sum += c[n]; } else c[n] = 0.f;
    }
    float inv = 1.f / sum;
    if (lane == 0) {
#pragma unroll 16
        for (int n = 0; n < 16; ++n)
            if (n < nn) aout[((size_t)k * m + i) * nn + n] = c[n] * inv;
    }
}

// ---------------------------------------------------------------------------
// x: [128,3,256] -> h0: rows=(b*256+node), cols 24 (3 channels tiled 8x)
// ---------------------------------------------------------------------------
__global__ void init_h(const float* __restrict__ x, float* __restrict__ h, int total)
{
    int t = blockIdx.x * blockDim.x + threadIdx.x;
    if (t >= total) return;
    int c = t % 24;
    int node = (t / 24) % 256;
    int b = t / (24 * 256);
    h[t] = x[((size_t)b * 3 + (c % 3)) * 256 + node];
}

__global__ void f32_to_f16(const float* __restrict__ w, _Float16* __restrict__ o, int n)
{
    int t = blockIdx.x * blockDim.x + threadIdx.x;
    if (t < n) o[t] = (_Float16)w[t];
}

// ---------------------------------------------------------------------------
// Sparse neighbour aggregations (heads stored [24, m, nn]; L1=3g, L2a=3g+1,
// L2b=3g+2). Column index recovered from flat idx via %m.
// ---------------------------------------------------------------------------
__global__ void agg_one_hop(const float* __restrict__ a, const int* __restrict__ idx,
                            const float* __restrict__ h, _Float16* __restrict__ y,
                            int m, int nn, int C, int total)
{
    int t = blockIdx.x * blockDim.x + threadIdx.x;
    if (t >= total) return;
    int c = t % C;
    int g = (t / C) & 7;
    int row = t / (C * 8);
    int i = row % m;
    int b = row / m;
    const float* ar = a + ((size_t)(3 * g) * m + i) * nn;
    const int*  ir = idx + i * nn;
    float s = 0.f;
    for (int n = 0; n < nn; ++n) {
        int j = ir[n] % m;
        s += ar[n] * h[((size_t)b * m + j) * C + c];
    }
    y[(size_t)row * (8 * C) + g * C + c] = (_Float16)s;
}

__global__ void agg_two_hop_t(const float* __restrict__ a, const int* __restrict__ idx,
                              const float* __restrict__ h, _Float16* __restrict__ tb,
                              int m, int nn, int C, int total)
{
    int t = blockIdx.x * blockDim.x + threadIdx.x;
    if (t >= total) return;
    int c = t % C;
    int j = (t / C) % m;
    int b = (t / (C * m)) % 128;
    int g = t / (C * m * 128);
    const float* ar = a + ((size_t)(3 * g + 1) * m + j) * nn;
    const int*  ir = idx + j * nn;
    float s = 0.f;
    for (int n = 0; n < nn; ++n) {
        int jj = ir[n] % m;
        s += ar[n] * h[((size_t)b * m + jj) * C + c];
    }
    tb[(((size_t)g * 128 + b) * m + j) * C + c] = (_Float16)s;
}

__global__ void agg_two_hop_y(const float* __restrict__ a, const int* __restrict__ idx,
                              const _Float16* __restrict__ tb, _Float16* __restrict__ y,
                              int m, int nn, int C, int total)
{
    int t = blockIdx.x * blockDim.x + threadIdx.x;
    if (t >= total) return;
    int c = t % C;
    int g = (t / C) & 7;
    int row = t / (C * 8);
    int i = row % m;
    int b = row / m;
    const float* ar = a + ((size_t)(3 * g + 2) * m + i) * nn;
    const int*  ir = idx + i * nn;
    float s = 0.f;
    for (int n = 0; n < nn; ++n) {
        int j = ir[n] % m;
        s += ar[n] * (float)tb[(((size_t)g * 128 + b) * m + j) * C + c];
    }
    y[(size_t)row * (8 * C) + g * C + c] = (_Float16)s;
}

// ---------------------------------------------------------------------------
// WMMA GEMM: C[M,N] = A[M,K] @ B[K,N] + bias (optional relu).
// 4 waves / block; block tile 32 (rows) x 64 (cols); wave w owns a 32x16
// column slice => 2 accumulators sharing one B fragment (2 wmma / k-step).
// A tile (32x32) row-major in LDS; B tile (32x64) stored TRANSPOSED in LDS
// so both A and B fragment element-pairs are contiguous dwords
// (ds_load_2addr_b32 per pair, bit_cast to v16h -> no b16 moves).
// B global reads are 8-contiguous-half (16B) vector loads, scattered into
// the transposed tile, so global traffic pipelines instead of chaining.
// ---------------------------------------------------------------------------
template <bool A_F16>
__global__ void gemm_wmma(const void* __restrict__ Aptr, int lda,
                          const _Float16* __restrict__ Bw, const float* __restrict__ bias,
                          float* __restrict__ C, int ldc,
                          int M, int N, int K, int relu)
{
    __shared__ _Float16 As[32][34];   // 32 rows x 32 K (+pad)
    __shared__ _Float16 Bt[64][34];   // 64 cols x 32 K (+pad), transposed
    const int tid  = threadIdx.x;
    const int lane = tid & 31;
    const int wave = tid >> 5;
    const int rowBase  = blockIdx.x * 32;
    const int colBlock = blockIdx.y * 64;

    v8f acc0 = {}, acc1 = {};
    const int rrow   = lane & 15;
    const int kupper = (lane >> 4) << 3;   // +8 K-offset (and +8 M rows in D)
    const int wcol   = wave * 16 + rrow;
    const bool fullN = (colBlock + 64) <= N;

    for (int k0 = 0; k0 < K; k0 += 32) {
        // cooperative A tile load (32x32): 8 consecutive halfs per thread
#pragma unroll
        for (int u = 0; u < 8; ++u) {
            int id = tid * 8 + u;
            int r = id >> 5, kk = id & 31;
            _Float16 v;
            if (A_F16)
                v = ((const _Float16*)Aptr)[(size_t)(rowBase + r) * lda + k0 + kk];
            else
                v = (_Float16)((const float*)Aptr)[(size_t)(rowBase + r) * lda + k0 + kk];
            As[r][kk] = v;
        }
        // cooperative B tile load (32x64) -> transposed into Bt[col][K].
        // Full-tile path: each thread does two 16B vector loads of 8
        // contiguous halfs from one row, then scatters into Bt.
        if (fullN) {
#pragma unroll
            for (int hseg = 0; hseg < 2; ++hseg) {
                int id = (hseg * 128 + tid) * 8;      // 0..2040, step 8
                int r = id >> 6, cc = id & 63;
                v8h vv = *(const v8h*)&Bw[(size_t)(k0 + r) * N + colBlock + cc];
#pragma unroll
                for (int e = 0; e < 8; ++e)
                    Bt[cc + e][r] = vv[e];
            }
        } else {
#pragma unroll
            for (int j = 0; j < 16; ++j) {
                int id = j * 128 + tid;
                int r = id >> 6, cc = id & 63;
                int col = colBlock + cc;
                Bt[cc][r] = (col < N) ? Bw[(size_t)(k0 + r) * N + col] : (_Float16)0.f;
            }
        }
        __syncthreads();

        // fragments as 8 dword LDS loads each (pairs are contiguous)
        v8u au0, au1, bu;
#pragma unroll
        for (int p = 0; p < 8; ++p) {
            int kk = ((p & 3) * 2) + ((p >= 4) ? 16 : 0) + kupper;  // even
            au0[p] = *(const unsigned int*)&As[rrow][kk];
            au1[p] = *(const unsigned int*)&As[16 + rrow][kk];
            bu[p]  = *(const unsigned int*)&Bt[wcol][kk];
        }
        v16h af0 = __builtin_bit_cast(v16h, au0);
        v16h af1 = __builtin_bit_cast(v16h, au1);
        v16h bf  = __builtin_bit_cast(v16h, bu);
        acc0 = __builtin_amdgcn_wmma_f32_16x16x32_f16(false, af0, false, bf,
                                                      (short)0, acc0, false, false);
        acc1 = __builtin_amdgcn_wmma_f32_16x16x32_f16(false, af1, false, bf,
                                                      (short)0, acc1, false, false);
        __syncthreads();
    }

    // D layout: VGPR r -> M = r + 8*(lane>=16), N = lane&15
    const int ncol = colBlock + wave * 16 + rrow;
    if (ncol < N) {
        float bv = bias ? bias[ncol] : 0.f;
#pragma unroll
        for (int r = 0; r < 8; ++r) {
            int m0 = rowBase + r + kupper;
            float v0 = acc0[r] + bv;
            if (relu) v0 = fmaxf(v0, 0.f);
            C[(size_t)m0 * ldc + ncol] = v0;
            float v1 = acc1[r] + bv;
            if (relu) v1 = fmaxf(v1, 0.f);
            C[(size_t)(m0 + 16) * ldc + ncol] = v1;
        }
    }
}

// ---------------------------------------------------------------------------
// BatchNorm statistics over rows (B*m) for each channel.
// ---------------------------------------------------------------------------
__global__ void bn_stats(const float* __restrict__ h, int rows, int Cdim,
                         float* __restrict__ mean, float* __restrict__ var)
{
    const int ch = blockIdx.x;
    __shared__ float s1[256], s2[256];
    float a = 0.f, b = 0.f;
    for (int r = threadIdx.x; r < rows; r += 256) {
        float v = h[(size_t)r * Cdim + ch];
        a += v; b += v * v;
    }
    s1[threadIdx.x] = a; s2[threadIdx.x] = b;
    __syncthreads();
    for (int o = 128; o > 0; o >>= 1) {
        if (threadIdx.x < (unsigned)o) {
            s1[threadIdx.x] += s1[threadIdx.x + o];
            s2[threadIdx.x] += s2[threadIdx.x + o];
        }
        __syncthreads();
    }
    if (threadIdx.x == 0) {
        float mu = s1[0] / rows;
        mean[ch] = mu;
        var[ch]  = s2[0] / rows - mu * mu;
    }
}

__global__ void bn_apply_pool(const float* __restrict__ h, const int* __restrict__ pool,
                              const float* __restrict__ mean, const float* __restrict__ var,
                              const float* __restrict__ gg, const float* __restrict__ bb,
                              float* __restrict__ out,
                              int m_in, int m_out, int Cdim, int total, int do_pool)
{
    int t = blockIdx.x * blockDim.x + threadIdx.x;
    if (t >= total) return;
    int ch = t % Cdim;
    int p  = (t / Cdim) % m_out;
    int b  = t / (Cdim * m_out);
    float mu = mean[ch], iv = rsqrtf(var[ch] + BN_EPS);
    float gv = gg[ch], bv = bb[ch];
    float best;
    if (do_pool) {
        best = 0.f;  // values are relu'd (>= 0)
        for (int u = 0; u < 4; ++u) {
            int node = pool[p * 4 + u];
            float v = (h[((size_t)b * m_in + node) * Cdim + ch] - mu) * iv * gv + bv;
            best = fmaxf(best, fmaxf(v, 0.f));
        }
    } else {
        float v = (h[((size_t)b * m_in + p) * Cdim + ch] - mu) * iv * gv + bv;
        best = fmaxf(v, 0.f);
    }
    out[((size_t)b * m_out + p) * Cdim + ch] = best;
}

// ---------------------------------------------------------------------------
extern "C" void kernel_launch(void* const* d_in, const int* in_sizes, int n_in,
                              void* d_out, int out_size, void* d_ws, size_t ws_size,
                              hipStream_t stream)
{
    (void)in_sizes; (void)n_in; (void)out_size;

    // ---- inputs (setup_inputs dict order; lists flattened element-wise) ----
    const float* x = (const float*)d_in[0];
    const float* maps[4]; const int* idxs[4]; const int* pools[4];
    for (int lv = 0; lv < 4; ++lv) {
        maps[lv]  = (const float*)d_in[1 + lv];
        idxs[lv]  = (const int*)  d_in[5 + lv];
        pools[lv] = (const int*)  d_in[9 + lv];
    }
    const float* Lw1 = (const float*)d_in[13];
    const float* Lb1 = (const float*)d_in[14];
    const float* Lw2 = (const float*)d_in[15];
    const float* Lb2 = (const float*)d_in[16];
    const float* ince_w[16]; const float* ince_b[16];
    for (int j = 0; j < 16; ++j) {
        ince_w[j] = (const float*)d_in[17 + j];
        ince_b[j] = (const float*)d_in[33 + j];
    }
    const float* bng[8]; const float* bnb[8];
    for (int i = 0; i < 8; ++i) { bng[i] = (const float*)d_in[49 + i]; bnb[i] = (const float*)d_in[57 + i]; }
    const float* fc1w = (const float*)d_in[65];
    const float* fc1b = (const float*)d_in[66];
    const float* fc2w = (const float*)d_in[67];
    const float* fc2b = (const float*)d_in[68];

    // ---- workspace carve ----
    char* ws = (char*)d_ws;
    size_t off = 0;
    auto carve = [&](size_t bytes) -> void* {
        void* p = ws + off;
        off += (bytes + 255) & ~(size_t)255;
        return p;
    };
    float*    a_buf = (float*)   carve((size_t)24 * 256 * 16 * 4);         // sparse attention
    float*    h_a   = (float*)   carve((size_t)32768 * 256 * 4);           // h (ping)
    float*    h_b   = (float*)   carve((size_t)32768 * 256 * 4);           // h (pong / gemm out)
    _Float16* y_buf = (_Float16*)carve((size_t)32768 * 2048 * 2);          // aggregated feats (f16)
    _Float16* t_buf = (_Float16*)carve((size_t)8 * 128 * 256 * 256 * 2);   // two-hop temp (f16)
    _Float16* w16   = (_Float16*)carve((size_t)8192 * 512 * 2);            // f16 weight staging
    float*    meanb = (float*)   carve(1024 * 4);
    float*    varb  = (float*)   carve(1024 * 4);
    if (ws_size < off) return;   // workspace too small: leave output untouched

    const int L_CFGS[8]   = {0, 0, 1, 1, 2, 2, 3, 3};
    const int CFGS[8]     = {256, 256, 512, 512, 512, 512, 1024, 1024};
    const int POOLL[8]    = {-1, 0, -1, 1, -1, 2, -1, 3};
    const int LEVEL_NN[4] = {16, 12, 8, 3};

    { // h0 = tile(transpose(x)), rows = b*256+node, 24 channels
        int total = 128 * 256 * 24;
        init_h<<<(total + 255) / 256, 256, 0, stream>>>(x, h_a, total);
    }

    int m = 256, inC = 24;
    for (int i = 0; i < 8; ++i) {
        const int lv = L_CFGS[i], nn = LEVEL_NN[lv];
        const int outh = CFGS[i] / 2, Cfg = CFGS[i];
        const int Kdim = 8 * inC, M = 128 * m;

        // attention for this layer's 24 heads
        attn_kernel<<<24 * m, 32, 0, stream>>>(
            maps[lv],
            Lw1 + (size_t)(24 * i) * 5 * 128, Lb1 + (size_t)(24 * i) * 128,
            Lw2 + (size_t)(24 * i) * 128,     Lb2 + 24 * i,
            a_buf, m, nn);

        // one-hop aggregate -> y (f16), then WMMA projection -> h_b cols [0,outh)
        int tot1 = M * 8 * inC;
        agg_one_hop<<<(tot1 + 255) / 256, 256, 0, stream>>>(a_buf, idxs[lv], h_a, y_buf, m, nn, inC, tot1);
        int nW = Kdim * outh;
        f32_to_f16<<<(nW + 255) / 256, 256, 0, stream>>>(ince_w[2 * i], w16, nW);
        dim3 g1(M / 32, (outh + 63) / 64);
        gemm_wmma<true><<<g1, 128, 0, stream>>>(y_buf, Kdim, w16, ince_b[2 * i], h_b, Cfg, M, outh, Kdim, 0);

        // two-hop: t = L2a.h, y2 = L2b.t, project -> h_b cols [outh, 2*outh)
        int tott = 8 * 128 * m * inC;
        agg_two_hop_t<<<(tott + 255) / 256, 256, 0, stream>>>(a_buf, idxs[lv], h_a, t_buf, m, nn, inC, tott);
        agg_two_hop_y<<<(tot1 + 255) / 256, 256, 0, stream>>>(a_buf, idxs[lv], t_buf, y_buf, m, nn, inC, tot1);
        f32_to_f16<<<(nW + 255) / 256, 256, 0, stream>>>(ince_w[2 * i + 1], w16, nW);
        gemm_wmma<true><<<g1, 128, 0, stream>>>(y_buf, Kdim, w16, ince_b[2 * i + 1], h_b + outh, Cfg, M, outh, Kdim, 0);

        // batchnorm (stats over B*m rows), relu, optional max-pool-by-4 -> h_a
        bn_stats<<<Cfg, 256, 0, stream>>>(h_b, M, Cfg, meanb, varb);
        int mo = (POOLL[i] >= 0) ? m / 4 : m;
        int tot2 = 128 * mo * Cfg;
        bn_apply_pool<<<(tot2 + 255) / 256, 256, 0, stream>>>(
            h_b, (POOLL[i] >= 0) ? pools[POOLL[i]] : (const int*)nullptr,
            meanb, varb, bng[i], bnb[i], h_a, m, mo, Cfg, tot2, POOLL[i] >= 0 ? 1 : 0);

        m = mo; inC = Cfg;
    }

    // ---- final FCs (h_a is [128, 1024]) ----
    {
        int n = 1024 * 512;
        f32_to_f16<<<(n + 255) / 256, 256, 0, stream>>>(fc1w, w16, n);
        dim3 g(128 / 32, (512 + 63) / 64);
        gemm_wmma<false><<<g, 128, 0, stream>>>(h_a, 1024, w16, fc1b, h_b, 512, 128, 512, 1024, 1);
    }
    {
        int n = 512 * 10;
        f32_to_f16<<<(n + 255) / 256, 256, 0, stream>>>(fc2w, w16, n);
        dim3 g(128 / 32, 1);
        gemm_wmma<false><<<g, 128, 0, stream>>>(h_b, 512, w16, fc2b, (float*)d_out, 10, 128, 10, 512, 0);
    }
}
